// NeuralFSM_22179211117291
// MI455X (gfx1250) — compile-verified, matrix operand order checked by reading, not run on hardware
//
#include <hip/hip_runtime.h>
#include <hip/hip_bf16.h>

typedef __attribute__((ext_vector_type(16))) _Float16 v16h;
typedef __attribute__((ext_vector_type(8)))  float    v8f;

// -------------------------------------------------------------------------
// Kernel 1: rows [R x 8] float32, out[r] = (uint8) sum_t t * rows[r][t]
// (rows are one-hot / one-hot-selected, so the dot with [0..7] is an exact
//  integer in f16/f32). Done with V_WMMA_F32_16X16X32_F16: each wave handles
//  16 rows as the A matrix (K = 8 real + 24 zero), B's columns all hold the
//  weight vector [0..7], so C[:,n] is identical for every n.
// A layout (16-bit, 16x32): lanes 0-15 hold K=0..7 in v16h elems 0..7 (M=lane);
// lanes 16-31 hold K=8..15/24..31 -> all zero here.
// B layout (32x16): lanes 0-15 elems 0..7 are K=0..7 at N=lane; lanes 16-31
// are K=8..15/24..31 -> zero.
// C layout: VGPR j @ lanes 0-15 = C[M=j][N=lane]; @ lanes 16-31 = C[M=j+8].
// -------------------------------------------------------------------------
__global__ void fsm_dot8_u8_wmma(const float* __restrict__ rows,
                                 unsigned char* __restrict__ out, int R) {
    const int lane = threadIdx.x & 31;
    const int wid  = threadIdx.x >> 5;
    const int rowbase = (blockIdx.x * (blockDim.x >> 5) + wid) * 16;

    v16h a = {};
    if (lane < 16) {
        int r = rowbase + lane;
        if (r >= R) r = R - 1;              // clamped safe load; store is guarded
        const float* p = rows + (size_t)r * 8;
#pragma unroll
        for (int k = 0; k < 8; ++k) a[k] = (_Float16)p[k];
    }
    v16h b = {};
    if (lane < 16) {
#pragma unroll
        for (int k = 0; k < 8; ++k) b[k] = (_Float16)(float)k;
    }

    v8f c = {};
    c = __builtin_amdgcn_wmma_f32_16x16x32_f16(
        /*neg_a=*/false, a, /*neg_b=*/false, b,
        /*c_mod=*/(short)0, c, /*reuse_a=*/false, /*reuse_b=*/false);

    // lane l in [0,8): row rowbase+l is c[l];  lane l in [16,24): row rowbase+8+(l-16)
    int r = -1;
    int sel = 0;
    if (lane < 8)                      { r = rowbase + lane;          sel = lane; }
    else if (lane >= 16 && lane < 24)  { r = rowbase + 8 + (lane-16); sel = lane - 16; }
    float v = 0.0f;
#pragma unroll
    for (int j = 0; j < 8; ++j) v = (sel == j) ? c[j] : v;
    if (r >= 0 && r < R) out[r] = (unsigned char)(v + 0.5f);
}

// -------------------------------------------------------------------------
// Kernel 2: zero the neighborhood-mask array
// -------------------------------------------------------------------------
__global__ void fsm_zero_mask(unsigned int* __restrict__ mask, int N) {
    int n = blockIdx.x * blockDim.x + threadIdx.x;
    if (n < N) mask[n] = 0u;
}

// -------------------------------------------------------------------------
// Kernel 3: per edge:  mask[dst] |= 1 << state[src]
// Bitwise OR is commutative/idempotent -> deterministic regardless of atomic
// ordering. Edge list streamed with 128-bit loads (L2-resident: 51.2MB<192MB).
// -------------------------------------------------------------------------
__global__ void fsm_edge_scatter(const int* __restrict__ src,
                                 const int* __restrict__ dst,
                                 const unsigned char* __restrict__ state,
                                 unsigned int* __restrict__ mask, int E) {
    int t = blockIdx.x * blockDim.x + threadIdx.x;
    int base = t * 4;
    if (base + 3 < E) {
        int4 s4 = *(const int4*)(src + base);
        int4 d4 = *(const int4*)(dst + base);
        unsigned int b0 = 1u << state[s4.x];
        unsigned int b1 = 1u << state[s4.y];
        unsigned int b2 = 1u << state[s4.z];
        unsigned int b3 = 1u << state[s4.w];
        atomicOr(&mask[d4.x], b0);
        atomicOr(&mask[d4.y], b1);
        atomicOr(&mask[d4.z], b2);
        atomicOr(&mask[d4.w], b3);
    } else {
        for (int e = base; e < E; ++e) {
            unsigned int b = 1u << state[src[e]];
            atomicOr(&mask[dst[e]], b);
        }
    }
}

// -------------------------------------------------------------------------
// Kernel 4: state[n] = nexttab[mask[n]*8 + state[n]]; re-zero mask for the
// next iteration (separate launches order scatter vs. update).
// -------------------------------------------------------------------------
__global__ void fsm_update_state(unsigned char* __restrict__ state,
                                 unsigned int* __restrict__ mask,
                                 const unsigned char* __restrict__ nexttab,
                                 int N) {
    int n = blockIdx.x * blockDim.x + threadIdx.x;
    if (n < N) {
        unsigned int m = mask[n];
        mask[n] = 0u;
        unsigned int s = state[n];
        state[n] = nexttab[m * 8u + s];
    }
}

// -------------------------------------------------------------------------
// Kernel 5: expand final uint8 state to one-hot float32 [N,8]
// -------------------------------------------------------------------------
__global__ void fsm_expand_onehot(const unsigned char* __restrict__ state,
                                  float* __restrict__ out, int N) {
    int n = blockIdx.x * blockDim.x + threadIdx.x;
    if (n >= N) return;
    int s = state[n];
    float4 lo = make_float4(s == 0 ? 1.f : 0.f, s == 1 ? 1.f : 0.f,
                            s == 2 ? 1.f : 0.f, s == 3 ? 1.f : 0.f);
    float4 hi = make_float4(s == 4 ? 1.f : 0.f, s == 5 ? 1.f : 0.f,
                            s == 6 ? 1.f : 0.f, s == 7 ? 1.f : 0.f);
    float4* o = (float4*)(out + (size_t)n * 8);
    o[0] = lo;
    o[1] = hi;
}

extern "C" void kernel_launch(void* const* d_in, const int* in_sizes, int n_in,
                              void* d_out, int out_size, void* d_ws, size_t ws_size,
                              hipStream_t stream) {
    const float* s0        = (const float*)d_in[0];  // [N, 8] one-hot f32
    const int*   edge_idx  = (const int*)  d_in[1];  // [2, E] int32
    const float* T         = (const float*)d_in[2];  // [256, 8, 8] f32 -> [2048, 8]

    const int N = in_sizes[0] / 8;
    const int E = in_sizes[1] / 2;
    const int RT = in_sizes[2] / 8;                  // 2048 rows of T
    const int ITERS = 20;

    const int* src = edge_idx;
    const int* dst = edge_idx + E;

    // workspace layout: mask [N u32] | state [N u8] | nexttab [RT u8]
    unsigned char* wsb   = (unsigned char*)d_ws;
    unsigned int*  mask  = (unsigned int*)wsb;
    size_t off = (size_t)N * sizeof(unsigned int);
    off = (off + 15) & ~(size_t)15;
    unsigned char* state = wsb + off;
    off += (size_t)N;
    off = (off + 15) & ~(size_t)15;
    unsigned char* nexttab = wsb + off;
    (void)ws_size; (void)n_in; (void)out_size;

    const int TPB = 128;                 // 4 waves -> 64 rows per block (WMMA kernels)
    const int rowsPerBlk = (TPB / 32) * 16;

    // 1) nexttab[x*8+s] = argmax_t T[x,s,t]  (via WMMA dot with [0..7])
    fsm_dot8_u8_wmma<<<(RT + rowsPerBlk - 1) / rowsPerBlk, TPB, 0, stream>>>(T, nexttab, RT);
    // 2) state[n] = argmax_s s0[n,s]         (via WMMA dot with [0..7])
    fsm_dot8_u8_wmma<<<(N + rowsPerBlk - 1) / rowsPerBlk, TPB, 0, stream>>>(s0, state, N);
    // 3) clear masks
    fsm_zero_mask<<<(N + 255) / 256, 256, 0, stream>>>(mask, N);

    // 4) 20 iterations of scatter-OR + table transition
    const int nVec   = (E + 3) / 4;
    const int gEdge  = (nVec + 255) / 256;
    const int gNode  = (N + 255) / 256;
    for (int it = 0; it < ITERS; ++it) {
        fsm_edge_scatter<<<gEdge, 256, 0, stream>>>(src, dst, state, mask, E);
        fsm_update_state<<<gNode, 256, 0, stream>>>(state, mask, nexttab, N);
    }

    // 5) one-hot expand to float32 output [N, 8]
    fsm_expand_onehot<<<gNode, 256, 0, stream>>>(state, (float*)d_out, N);
}
